// SSMNet_46188078301338
// MI455X (gfx1250) — matrix operand order repaired and checked by reading, not compiled, not run on hardware
//
#include <hip/hip_runtime.h>

typedef __attribute__((ext_vector_type(16))) _Float16 v16h;
typedef __attribute__((ext_vector_type(8)))  float    v8f;

union V16H {
    v16h     v;
    uint4    q[2];
    _Float16 h[16];
};

#define T_LEN   1024
#define SAUG_W  168   // 128 (S2) + 16 (X) + 16 (zero pad to K=160) + 8 (bank pad)
#define NCONS   4     // consumer waves; each owns 2 N-tiles of 16 cols
#define NTHREADS (32 * (NCONS + 1))

__global__ __launch_bounds__(NTHREADS)
void ssm_fused_kernel(const float* __restrict__ u,
                      const float* __restrict__ A1,
                      const float* __restrict__ B1,
                      const float* __restrict__ C1,
                      const float* __restrict__ A2,
                      const float* __restrict__ B2,
                      const float* __restrict__ C2,
                      const float* __restrict__ W,
                      const float* __restrict__ bias,
                      float* __restrict__ out)
{
    // Double-buffered augmented state [S2 (128) | X (16) | zeros], f16, row-major.
    __shared__ _Float16 Saug[2][16][SAUG_W];
    // Layer-1 state, 16x16 padded to K=32 (cols 16..31 stay zero).
    __shared__ _Float16 S1buf[16][32];
    // Epilogue staging.
    __shared__ float stage[16][132];
    __shared__ float vstage[16][132];

    const int tid  = threadIdx.x;
    const int wave = tid >> 5;      // waves 0..3: layer-2 (2 tiles each); wave 4: producer
    const int lane = tid & 31;
    const int lh   = lane >> 4;     // lane half
    const int lc   = lane & 15;     // lane column / row index
    const int b0   = blockIdx.x * 16;

    // ---------------- zero-init shared state ----------------
    {
        _Float16* p = &Saug[0][0][0];
        for (int i = tid; i < 2 * 16 * SAUG_W; i += NTHREADS) p[i] = (_Float16)0.f;
        _Float16* q = &S1buf[0][0];
        for (int i = tid; i < 16 * 32; i += NTHREADS) q[i] = (_Float16)0.f;
    }

    const v8f zero8 = {};

    // ---------------- weight preload into B-operand lane layout ----------------
    // B operand as packed here (contiguous k, 16-bit, KxN=32x16): lane n<16 holds col n, K=0..15 (2/VGPR);
    // lanes 16-31 hold col n-16, K=16..31.
    v16h bw0[5], bw1[5];            // [[A2],[B2],[0]] columns for this wave's 2 N-tiles
    v16h vA1 = {}, vP = {};         // layer-1: A1 and P = A1@C1
    float b1n = 0.f, rn = 0.f;      // B1[col], (B1@C1)[col]
    float ucur[8];                  // software-pipelined u values (producer only)

    if (wave < NCONS) {
#pragma unroll
        for (int tile = 0; tile < 2; ++tile) {
            const int col = (2 * wave + tile) * 16 + lc;
#pragma unroll
            for (int c = 0; c < 5; ++c) {
                V16H t;
#pragma unroll
                for (int e = 0; e < 16; ++e) {
                    const int k = 32 * c + 16 * lh + e;   // augmented K index
                    float v = 0.f;
                    if (k < 128)      v = A2[k * 128 + col];
                    else if (k < 144) v = B2[(k - 128) * 128 + col];
                    t.h[e] = (_Float16)v;
                }
                if (tile == 0) bw0[c] = t.v; else bw1[c] = t.v;
            }
        }
#pragma unroll
        for (int v = 0; v < 8; ++v) ucur[v] = 0.f;
    } else {
        b1n = B1[lc];
        float racc = 0.f;
        for (int k = 0; k < 16; ++k) racc += B1[k] * C1[k * 16 + lc];
        rn = racc;
        V16H ta, tp;
#pragma unroll
        for (int e = 0; e < 16; ++e) {
            float va = 0.f, vp = 0.f;
            if (lh == 0) {                 // K=0..15 real rows, K=16..31 zero pad
                va = A1[e * 16 + lc];
                float s = 0.f;             // P[e][lc] = sum_k A1[e][k] * C1[k][lc]
                for (int k = 0; k < 16; ++k) s += A1[e * 16 + k] * C1[k * 16 + lc];
                vp = s;
            }
            ta.h[e] = (_Float16)va;
            tp.h[e] = (_Float16)vp;
        }
        vA1 = ta.v;
        vP  = tp.v;
#pragma unroll
        for (int c = 0; c < 5; ++c) { bw0[c] = vA1; bw1[c] = vA1; }  // unused
#pragma unroll
        for (int v = 0; v < 8; ++v)        // preload u for step 0
            ucur[v] = u[(size_t)(b0 + v + 8 * lh) * T_LEN];
    }

    // Producer: one step of layer 1 using s_{t-1} only (P-trick), deposit X_t.
    // s_t = s_{t-1}@A1 + u_t(x)B1 ; y_t = s_{t-1}@P + u_t(x)r ; X_t = softsign(y_t)
    auto producer_step = [&](int tt, int wb) {
        float unx[8];                       // prefetch u for next step (off critical path)
        const int tn = (tt + 1 < T_LEN) ? tt + 1 : tt;
#pragma unroll
        for (int v = 0; v < 8; ++v)
            unx[v] = u[(size_t)(b0 + v + 8 * lh) * T_LEN + tn];

        V16H a;                             // s_{t-1} as A-operand
        {
            const _Float16* base = &S1buf[lc][16 * lh];
            a.q[0] = *(const uint4*)(base);
            a.q[1] = *(const uint4*)(base + 8);
        }
        v8f dsn = zero8, dy = zero8;        // two independent WMMAs on same A
        dsn = __builtin_amdgcn_wmma_f32_16x16x32_f16(false, a.v, false, vA1,
                                                     (short)0, dsn, false, false);
        dy  = __builtin_amdgcn_wmma_f32_16x16x32_f16(false, a.v, false, vP,
                                                     (short)0, dy, false, false);
#pragma unroll
        for (int v = 0; v < 8; ++v) {       // s_t -> S1buf (C/D layout scatter)
            const float s = dsn[v] + ucur[v] * b1n;
            S1buf[v + 8 * lh][lc] = (_Float16)s;
        }
#pragma unroll
        for (int v = 0; v < 8; ++v) {       // X_t -> augmented-state region
            float y = dy[v] + ucur[v] * rn;
            y = y * rsqrtf(1.f + y * y);    // softsign
            Saug[wb][v + 8 * lh][128 + lc] = (_Float16)y;
        }
#pragma unroll
        for (int v = 0; v < 8; ++v) ucur[v] = unx[v];
    };

    __syncthreads();                        // shared zero-init visible
    if (wave == NCONS) producer_step(0, 0); // X_0 into buffer 0 (S2_0 = 0 already)
    __syncthreads();

    // ---------------- main recurrence: 1024 steps, 1 barrier each ----------------
    v8f acc0 = zero8, acc1 = zero8;         // consumer: last S2 tiles (f32)
    for (int t = 0; t < T_LEN; ++t) {
        const int rb = t & 1;
        const int wb = rb ^ 1;
        if (wave < NCONS) {
            v8f r0 = zero8, r1 = zero8;
#pragma unroll
            for (int c = 0; c < 5; ++c) {   // K = 160 in 5 chunks of 32
                V16H a;                     // shared A feeds two N-tiles
                const _Float16* base = &Saug[rb][lc][32 * c + 16 * lh];
                a.q[0] = *(const uint4*)(base);
                a.q[1] = *(const uint4*)(base + 8);
                r0 = __builtin_amdgcn_wmma_f32_16x16x32_f16(false, a.v, false, bw0[c],
                                                            (short)0, r0, false, false);
                r1 = __builtin_amdgcn_wmma_f32_16x16x32_f16(false, a.v, false, bw1[c],
                                                            (short)0, r1, false, false);
            }
#pragma unroll
            for (int v = 0; v < 8; ++v) {   // write S2_{t+1} tiles into next buffer
                Saug[wb][v + 8 * lh][(2 * wave) * 16 + lc]     = (_Float16)r0[v];
                Saug[wb][v + 8 * lh][(2 * wave + 1) * 16 + lc] = (_Float16)r1[v];
            }
            acc0 = r0;
            acc1 = r1;
        } else if (t + 1 < T_LEN) {
            producer_step(t + 1, wb);       // X_{t+1}, pipelined one step ahead
        }
        __syncthreads();
    }

    // ---------------- epilogue: out = softsign(S2_T @ C2) @ W + b ----------------
    if (wave < NCONS) {
#pragma unroll
        for (int v = 0; v < 8; ++v) {
            stage[v + 8 * lh][(2 * wave) * 16 + lc]     = acc0[v];
            stage[v + 8 * lh][(2 * wave + 1) * 16 + lc] = acc1[v];
        }
    }
    __syncthreads();

    for (int idx = tid; idx < 16 * 128; idx += NTHREADS) {
        const int row = idx >> 7;
        const int j   = idx & 127;
        float s = 0.f;
        for (int k = 0; k < 128; ++k) s += stage[row][k] * C2[k * 128 + j];
        s = s * rsqrtf(1.f + s * s);
        vstage[row][j] = s;
    }
    __syncthreads();

    for (int idx = tid; idx < 16 * 10; idx += NTHREADS) {
        const int row = idx / 10;
        const int cls = idx - row * 10;
        float s = bias[cls];
        for (int j = 0; j < 128; ++j) s += vstage[row][j] * W[j * 10 + cls];
        out[(size_t)(b0 + row) * 10 + cls] = s;
    }
}

extern "C" void kernel_launch(void* const* d_in, const int* in_sizes, int n_in,
                              void* d_out, int out_size, void* d_ws, size_t ws_size,
                              hipStream_t stream) {
    const float* u  = (const float*)d_in[0];
    const float* A1 = (const float*)d_in[1];
    const float* B1 = (const float*)d_in[2];
    const float* C1 = (const float*)d_in[3];
    const float* A2 = (const float*)d_in[4];
    const float* B2 = (const float*)d_in[5];
    const float* C2 = (const float*)d_in[6];
    const float* W  = (const float*)d_in[7];
    const float* b  = (const float*)d_in[8];
    float* out = (float*)d_out;

    dim3 grid(16);          // 256 batch rows / 16 rows per workgroup
    dim3 block(NTHREADS);   // 4 consumer waves (2 N-tiles each) + 1 producer wave
    hipLaunchKernelGGL(ssm_fused_kernel, grid, block, 0, stream,
                       u, A1, B1, C1, A2, B2, C2, W, b, out);
    (void)in_sizes; (void)n_in; (void)out_size; (void)d_ws; (void)ws_size;
}
